// GPSConv_58171037057244
// MI455X (gfx1250) — matrix operand order repaired and checked by reading, not compiled, not run on hardware
//
#include <hip/hip_runtime.h>
#include <hip/hip_bf16.h>
#include <stdint.h>

// ---------------------------------------------------------------------------
// GPSConv layer for MI455X (gfx1250, wave32, WMMA).
// All GEMMs use v_wmma_f32_16x16x32_bf16 (bf16 in, f32 accum).
// Fused attention: scores(QK^T+bias+mask) -> softmax -> AV in one kernel,
// scores held in VGPRs, probs transposed through LDS into A-fragment layout.
// Problem constants (static per reference):
//   N=5120 nodes, C=128, d=16, B=32 graphs, M=256 dense slots, H=8 heads.
// ---------------------------------------------------------------------------

#define Nn 5120
#define Cc 128
#define Dd 16
#define Mm 256
#define Bb 32
#define Hh 8
#define EPSF 1e-8f

typedef __attribute__((ext_vector_type(16))) __bf16 v16bf;
typedef __attribute__((ext_vector_type(8)))  __bf16 v8bf;
typedef __attribute__((ext_vector_type(8)))  float  v8f;

__device__ inline v8f wmma_bf16(v16bf a, v16bf b, v8f c) {
  // D = A(16x32 bf16) * B(32x16 bf16) + C(16x16 f32)
  return __builtin_amdgcn_wmma_f32_16x16x32_bf16(
      /*neg_a=*/false, a, /*neg_b=*/false, b,
      /*c_mod=*/(short)0, c, /*reuse_a=*/false, /*reuse_b=*/false);
}

// Load 16 bf16 as two 8-element (16-byte) chunks at base+off0, base+off1.
// Works for both global and LDS pointers (offsets must be 16B aligned).
__device__ inline v16bf frag2(const __bf16* base, int off0, int off1) {
  v8bf lo = *(const v8bf*)(base + off0);
  v8bf hi = *(const v8bf*)(base + off1);
  v16bf r;
#pragma unroll
  for (int e = 0; e < 8; ++e) { r[e] = lo[e]; r[e + 8] = hi[e]; }
  return r;
}

// A-fragment (16x32 bf16) from row-major row pointer (at A + m*K + kc):
//   lanes 0-15: K = e (e<8), 16+(e-8) (e>=8); lanes 16-31: +8 -> ISA layout.
__device__ inline v16bf frag_a(const __bf16* rowKc, int hh) {
  return frag2(rowKc, 8 * hh, 16 + 8 * hh);
}
// B-fragment (32x16 bf16) from BT row pointer (at BT + n*K + kc): same K layout as frag_a.
__device__ inline v16bf frag_b(const __bf16* rowKc, int hh) {
  return frag2(rowKc, 8 * hh, 16 + 8 * hh);
}

// ---------------------------------------------------------------------------
// Small utility kernels
// ---------------------------------------------------------------------------
__global__ void k_zero(uint32_t* p, int ndw) {
  int i = blockIdx.x * blockDim.x + threadIdx.x;
  if (i < ndw) p[i] = 0u;
}

__global__ void k_f32_to_bf16(const float* s, __bf16* d, int n) {
  int i = blockIdx.x * blockDim.x + threadIdx.x;
  if (i < n) d[i] = (__bf16)s[i];
}

// offsets[b] = first index i with batch[i] >= b  (batch is sorted)
__global__ void k_offsets(const int* batch, int* offsets) {
  int b = threadIdx.x;
  if (b >= Bb) return;
  int lo = 0, hi = Nn;
  while (lo < hi) { int mid = (lo + hi) >> 1; if (batch[mid] < b) lo = mid + 1; else hi = mid; }
  offsets[b] = lo;
}

// one block (128 threads) per node: scatter into dense buffers
__global__ __launch_bounds__(128)
void k_scatter(const float* x, const float* per, const float* pei,
               const int* batch, const int* offsets,
               __bf16* hd, float* arD, float* aiD, float* maskD, int* intra) {
  int i = blockIdx.x, t = threadIdx.x;
  int b = batch[i];
  int ia = i - offsets[b];
  if (ia >= Mm) return;  // safety (reference assumes counts <= M)
  long row = (long)b * Mm + ia;
  hd[row * Cc + t] = (__bf16)x[(long)i * Cc + t];
  if (t < Dd) {
    arD[row * Dd + t] = per[(long)i * Dd + t];
    aiD[row * Dd + t] = pei[(long)i * Dd + t];
  }
  if (t == 0) { maskD[row] = 1.0f; intra[i] = ia; }
}

__global__ void k_pn(const float* arD, const float* aiD, float* pnD, int n) {
  int i = blockIdx.x * blockDim.x + threadIdx.x;
  if (i < n) {
    float a = arD[i], b = aiD[i];
    pnD[i] = sqrtf(a * a + b * b + EPSF);
  }
}

// dst[N][K] bf16 = transpose(src[K][N] f32)
__global__ void k_wT(const float* src, __bf16* dst, int K, int N) {
  int i = blockIdx.x * blockDim.x + threadIdx.x;
  if (i >= K * N) return;
  int n = i / K, k = i % K;
  dst[n * K + k] = (__bf16)src[k * N + n];
}

// pe_w2 [32][8] -> W2T [16][32] (cols padded 8..15 with zeros)
__global__ void k_w2pad(const float* src, __bf16* dst) {
  int i = blockIdx.x * blockDim.x + threadIdx.x;
  if (i >= 16 * 32) return;
  int n = i / 32, k = i % 32;
  dst[i] = (n < 8) ? (__bf16)src[k * 8 + n] : (__bf16)0.0f;
}

// v [B*M, C] f32 -> vT [B][H][16(dh)][M] bf16 (contiguous along j for B-frags)
__global__ void k_vT(const float* v, __bf16* vT) {
  int i = blockIdx.x * blockDim.x + threadIdx.x;
  if (i >= Bb * Mm * Cc) return;
  int c = i & (Cc - 1);
  int bm = i / Cc;
  int b = bm / Mm, j = bm % Mm;
  int h = c >> 4, d = c & 15;
  vT[(((long)b * Hh + h) * 16 + d) * Mm + j] = (__bf16)v[i];
}

// ---------------------------------------------------------------------------
// Generic WMMA GEMM: C[M,N] = A[M,K](bf16) * BT[N,K](bf16)^T + bias (+ReLU)
// One wave per 16x16 output tile. M,N multiples of 16, K multiple of 32.
// ---------------------------------------------------------------------------
__global__ __launch_bounds__(32)
void k_gemm(const __bf16* A, const __bf16* BT, const float* bias,
            float* Cout, int Mr, int Ncols, int K, int relu) {
  int tiles_n = Ncols >> 4;
  int id = blockIdx.x;
  int tn = id % tiles_n, tm = id / tiles_n;
  int lane = threadIdx.x, m = lane & 15, hh = lane >> 4;
  const __bf16* arow = A + (long)(tm * 16 + m) * K;
  const __bf16* brow = BT + (long)(tn * 16 + m) * K;
  v8f acc = {0.f, 0.f, 0.f, 0.f, 0.f, 0.f, 0.f, 0.f};
  for (int kc = 0; kc < K; kc += 32) {
    v16bf a = frag_a(arow + kc, hh);
    v16bf b = frag_b(brow + kc, hh);
    acc = wmma_bf16(a, b, acc);
  }
  int col = tn * 16 + m;
  float bb = bias ? bias[col] : 0.f;
#pragma unroll
  for (int r = 0; r < 8; ++r) {
    int row = tm * 16 + r + 8 * hh;
    float v = acc[r] + bb;
    if (relu) v = fmaxf(v, 0.f);
    Cout[(long)row * Ncols + col] = v;
  }
}

// ---------------------------------------------------------------------------
// Fused pairwise PE-bias MLP. One wave per (b, P, q-tile of 16 Q's).
// Builds bias_in tile [16 pairs x 64 feats] in LDS, runs 64->32 (ReLU) -> 8
// via 5 WMMAs, stores into biasArr[b][h][i=Q][j=P] (the reference transpose).
// ---------------------------------------------------------------------------
__global__ __launch_bounds__(32)
void k_pair(const float* arD, const float* aiD, const float* pnD,
            const __bf16* w1T, const __bf16* w2T,
            const float* pe_b1, const float* pe_b2, float* biasArr) {
  __shared__ __bf16 Af[16][64];   // feature tile
  __shared__ __bf16 A2[16][32];   // hidden tile (post-ReLU)

  int id = blockIdx.x;              // Bb * Mm * 16 blocks
  int qt = id & 15;
  int P  = (id >> 4) & (Mm - 1);
  int b  = id >> 12;
  int q0 = qt * 16;
  int lane = threadIdx.x;

  const float* arB = arD + (long)b * Mm * Dd;
  const float* aiB = aiD + (long)b * Mm * Dd;
  const float* pnB = pnD + (long)b * Mm * Dd;

  // build 16x64 feature tile: [pn_P | pn_Q | real | imag]
  for (int idx = lane; idx < 16 * 64; idx += 32) {
    int r = idx >> 6, k = idx & 63;
    int Q = q0 + r;
    float val;
    if (k < 16) {
      val = pnB[P * Dd + k];
    } else if (k < 32) {
      val = pnB[Q * Dd + (k - 16)];
    } else {
      int kk = k & 15;
      float arP = arB[P * Dd + kk], aiP = aiB[P * Dd + kk];
      float arQ = arB[Q * Dd + kk], aiQ = aiB[Q * Dd + kk];
      float den = pnB[P * Dd + kk] * pnB[Q * Dd + kk] + EPSF;
      val = (k < 48) ? (arP * arQ + aiP * aiQ) / den
                     : (aiP * arQ - arP * aiQ) / den;
    }
    Af[r][k] = (__bf16)val;
  }
  __syncthreads();

  int m = lane & 15, hh = lane >> 4;

  // layer 1: [16x64] x [64x32] -> [16x32], ReLU, to LDS
  for (int nt = 0; nt < 2; ++nt) {
    v8f acc = {0.f, 0.f, 0.f, 0.f, 0.f, 0.f, 0.f, 0.f};
#pragma unroll
    for (int c = 0; c < 2; ++c) {
      v16bf a = frag_a(&Af[m][32 * c], hh);
      v16bf bf = frag_b(&w1T[(nt * 16 + m) * 64 + 32 * c], hh);
      acc = wmma_bf16(a, bf, acc);
    }
    float bb = pe_b1[nt * 16 + m];
#pragma unroll
    for (int r = 0; r < 8; ++r) {
      int row = r + 8 * hh;
      A2[row][nt * 16 + m] = (__bf16)fmaxf(acc[r] + bb, 0.f);
    }
  }
  __syncthreads();

  // layer 2: [16x32] x [32x16(pad of 8)] -> [16x16]
  v8f acc2 = {0.f, 0.f, 0.f, 0.f, 0.f, 0.f, 0.f, 0.f};
  {
    v16bf a = frag_a(&A2[m][0], hh);
    v16bf bf = frag_b(&w2T[m * 32], hh);
    acc2 = wmma_bf16(a, bf, acc2);
  }
  int h = m;
  if (h < Hh) {
    float bb2 = pe_b2[h];
#pragma unroll
    for (int r = 0; r < 8; ++r) {
      int Q = q0 + r + 8 * hh;
      // scores[b,h,i,j] consumes hb[b, j, i, h]  ->  store at [b][h][Q][P]
      biasArr[(((long)b * Hh + h) * Mm + Q) * Mm + P] = acc2[r] + bb2;
    }
  }
}

// ---------------------------------------------------------------------------
// Fused attention: one wave per (b, h, i-tile of 16 rows).
//   scores[i, 0..255] = QK^T/sqrt(16) + bias, key mask   (16 WMMAs, in VGPRs)
//   row softmax (shfl_xor reductions within each 16-lane half)
//   probs -> LDS (bf16, A-fragment layout) -> AV (8 WMMAs) -> o
// ---------------------------------------------------------------------------
__global__ __launch_bounds__(32)
void k_attn(const float* q, const float* kM, const float* maskD,
            const float* biasArr, const __bf16* vT, float* o) {
  __shared__ __bf16 probs[16][256];
  int id = blockIdx.x;               // Bb*Hh*16
  int ti = id & 15, bh = id >> 4;
  int b = bh >> 3, h = bh & 7;
  int lane = threadIdx.x, m = lane & 15, hh = lane >> 4;

  // Q fragment once (dh=16 padded to K=32 with zeros)
  v16bf aq;
  {
    const float* qp = q + ((long)b * Mm + ti * 16 + m) * Cc + h * 16 + 8 * hh;
#pragma unroll
    for (int e = 0; e < 8; ++e) { aq[e] = (__bf16)qp[e]; aq[e + 8] = (__bf16)0.f; }
  }

  // --- scores: 16 j-tiles, kept in registers (16 x v8f) ---
  v8f sc[16];
  const long bias_base = ((long)bh * Mm + ti * 16 + 8 * hh) * Mm;
#pragma unroll
  for (int t = 0; t < 16; ++t) {
    v16bf bf;
    const float* kp = kM + ((long)b * Mm + t * 16 + m) * Cc + h * 16;
#pragma unroll
    for (int e = 0; e < 16; ++e) bf[e] = (e < 8) ? (__bf16)kp[8 * hh + e] : (__bf16)0.f;
    v8f acc = {0.f, 0.f, 0.f, 0.f, 0.f, 0.f, 0.f, 0.f};
    acc = wmma_bf16(aq, bf, acc);

    int j = t * 16 + m;
    float mj = maskD[(long)b * Mm + j];
    const float* bp = biasArr + bias_base + j;
    if (t < 15) __builtin_prefetch(bp + 16, 0, 1);  // next bias tile -> global_prefetch_b8
#pragma unroll
    for (int r = 0; r < 8; ++r) {
      float s = acc[r] * 0.25f + bp[(long)r * Mm];  // 1/sqrt(dh)=0.25
      sc[t][r] = (mj > 0.5f) ? s : -1e9f;
    }
  }

  // --- softmax per row i = 8*hh + r (spread over 16 lanes of this half) ---
#pragma unroll
  for (int r = 0; r < 8; ++r) {
    float mx = -3.4e38f;
#pragma unroll
    for (int t = 0; t < 16; ++t) mx = fmaxf(mx, sc[t][r]);
#pragma unroll
    for (int off = 1; off < 16; off <<= 1) mx = fmaxf(mx, __shfl_xor(mx, off, 32));
    float s = 0.f;
#pragma unroll
    for (int t = 0; t < 16; ++t) { float e = __expf(sc[t][r] - mx); sc[t][r] = e; s += e; }
#pragma unroll
    for (int off = 1; off < 16; off <<= 1) s += __shfl_xor(s, off, 32);
    float inv = 1.f / s;
    int row = 8 * hh + r;
#pragma unroll
    for (int t = 0; t < 16; ++t) probs[row][t * 16 + m] = (__bf16)(sc[t][r] * inv);
  }
  __syncthreads();

  // --- AV: o[b,i,h,:] = sum_j probs * v ---
  const __bf16* vrow = vT + ((long)bh * 16 + m) * Mm;
  v8f accO = {0.f, 0.f, 0.f, 0.f, 0.f, 0.f, 0.f, 0.f};
#pragma unroll
  for (int jc = 0; jc < Mm; jc += 32) {
    v16bf a = frag_a(&probs[m][jc], hh);
    v16bf bf = frag_b(vrow + jc, hh);
    accO = wmma_bf16(a, bf, accO);
  }
#pragma unroll
  for (int r = 0; r < 8; ++r) {
    int i = ti * 16 + r + 8 * hh;
    o[((long)b * Mm + i) * Cc + h * 16 + m] = accO[r];
  }
}

// ---------------------------------------------------------------------------
// residual + LayerNorm; one 128-thread block per node row
// ---------------------------------------------------------------------------
__device__ inline float block_sum128(float v, float* red, int t) {
  red[t] = v; __syncthreads();
#pragma unroll
  for (int s = 64; s > 0; s >>= 1) { if (t < s) red[t] += red[t + s]; __syncthreads(); }
  float r = red[0]; __syncthreads();
  return r;
}

__global__ __launch_bounds__(128)
void k_resln(const float* oproj, const float* x, const int* batch, const int* intra,
             const float* g, const float* bta, float* hF, __bf16* hB) {
  __shared__ float red[128];
  int i = blockIdx.x, t = threadIdx.x;
  int b = batch[i], ia = intra[i];
  float v = oproj[((long)b * Mm + ia) * Cc + t] + x[(long)i * Cc + t];
  float mu = block_sum128(v, red, t) * (1.f / Cc);
  float c = v - mu;
  float var = block_sum128(c * c, red, t) * (1.f / Cc);
  float out = c * rsqrtf(var + 1e-5f) * g[t] + bta[t];
  hF[(long)i * Cc + t] = out;
  hB[(long)i * Cc + t] = (__bf16)out;
}

__global__ __launch_bounds__(128)
void k_final(const float* hF, const float* ffn2, const float* g, const float* bta,
             float* out) {
  __shared__ float red[128];
  int i = blockIdx.x, t = threadIdx.x;
  float v = hF[(long)i * Cc + t] + ffn2[(long)i * Cc + t];
  float mu = block_sum128(v, red, t) * (1.f / Cc);
  float c = v - mu;
  float var = block_sum128(c * c, red, t) * (1.f / Cc);
  out[(long)i * Cc + t] = c * rsqrtf(var + 1e-5f) * g[t] + bta[t];
}

// ---------------------------------------------------------------------------
// host launcher
// ---------------------------------------------------------------------------
extern "C" void kernel_launch(void* const* d_in, const int* in_sizes, int n_in,
                              void* d_out, int out_size, void* d_ws, size_t ws_size,
                              hipStream_t stream) {
  (void)in_sizes; (void)n_in; (void)out_size; (void)ws_size;
  const float* x      = (const float*)d_in[0];
  const float* pe_re  = (const float*)d_in[1];
  const float* pe_im  = (const float*)d_in[2];
  const float* wq     = (const float*)d_in[3];
  const float* bq     = (const float*)d_in[4];
  const float* wk     = (const float*)d_in[5];
  const float* bk     = (const float*)d_in[6];
  const float* wv     = (const float*)d_in[7];
  const float* bv     = (const float*)d_in[8];
  const float* wo     = (const float*)d_in[9];
  const float* bo     = (const float*)d_in[10];
  const float* pe_w1  = (const float*)d_in[11];
  const float* pe_b1  = (const float*)d_in[12];
  const float* pe_w2  = (const float*)d_in[13];
  const float* pe_b2  = (const float*)d_in[14];
  const float* mlp_w1 = (const float*)d_in[15];
  const float* mlp_b1 = (const float*)d_in[16];
  const float* mlp_w2 = (const float*)d_in[17];
  const float* mlp_b2 = (const float*)d_in[18];
  const float* ln2_g  = (const float*)d_in[19];
  const float* ln2_b  = (const float*)d_in[20];
  const float* ln3_g  = (const float*)d_in[21];
  const float* ln3_b  = (const float*)d_in[22];
  const int*   batch  = (const int*)d_in[23];
  // d_in[24] edge_index, d_in[25] nmax: unused by the reference computation
  float* outp = (float*)d_out;

  // ---- bump allocator over workspace (~105 MB total) ----
  char* p = (char*)d_ws;
  auto alloc = [&](size_t bytes) -> void* {
    void* r = (void*)p; p += (bytes + 255) & ~(size_t)255; return r;
  };
  int*     offs  = (int*)alloc(Bb * 4);
  int*     intra = (int*)alloc(Nn * 4);
  __bf16*  hd    = (__bf16*)alloc((size_t)Bb * Mm * Cc * 2);
  float*   arD   = (float*)alloc((size_t)Bb * Mm * Dd * 4);
  float*   aiD   = (float*)alloc((size_t)Bb * Mm * Dd * 4);
  float*   pnD   = (float*)alloc((size_t)Bb * Mm * Dd * 4);
  float*   maskD = (float*)alloc((size_t)Bb * Mm * 4);
  __bf16*  wqT   = (__bf16*)alloc((size_t)Cc * Cc * 2);
  __bf16*  wkT   = (__bf16*)alloc((size_t)Cc * Cc * 2);
  __bf16*  wvT   = (__bf16*)alloc((size_t)Cc * Cc * 2);
  __bf16*  woT   = (__bf16*)alloc((size_t)Cc * Cc * 2);
  __bf16*  w1T   = (__bf16*)alloc((size_t)32 * 64 * 2);
  __bf16*  w2T   = (__bf16*)alloc((size_t)16 * 32 * 2);
  __bf16*  m1T   = (__bf16*)alloc((size_t)256 * Cc * 2);
  __bf16*  m2T   = (__bf16*)alloc((size_t)Cc * 256 * 2);
  float*   qF    = (float*)alloc((size_t)Bb * Mm * Cc * 4);
  float*   kF    = (float*)alloc((size_t)Bb * Mm * Cc * 4);
  float*   vF    = (float*)alloc((size_t)Bb * Mm * Cc * 4);
  __bf16*  vT    = (__bf16*)alloc((size_t)Bb * Hh * 16 * Mm * 2);
  float*   biasA = (float*)alloc((size_t)Bb * Hh * Mm * Mm * 4);  // 64 MB
  float*   oF    = (float*)alloc((size_t)Bb * Mm * Cc * 4);
  __bf16*  oB    = (__bf16*)alloc((size_t)Bb * Mm * Cc * 2);
  float*   oPrj  = (float*)alloc((size_t)Bb * Mm * Cc * 4);
  float*   hF    = (float*)alloc((size_t)Nn * Cc * 4);
  __bf16*  hB    = (__bf16*)alloc((size_t)Nn * Cc * 2);
  float*   f1F   = (float*)alloc((size_t)Nn * 256 * 4);
  __bf16*  f1B   = (__bf16*)alloc((size_t)Nn * 256 * 2);
  float*   f2F   = (float*)alloc((size_t)Nn * Cc * 4);

  auto zdw = [&](void* ptr, int ndw) {
    k_zero<<<(ndw + 255) / 256, 256, 0, stream>>>((uint32_t*)ptr, ndw);
  };
  // zero scatter targets (ws is poisoned; must re-zero every call)
  zdw(hd, Bb * Mm * Cc / 2);
  zdw(arD, Bb * Mm * Dd);
  zdw(aiD, Bb * Mm * Dd);
  zdw(maskD, Bb * Mm);

  k_offsets<<<1, 32, 0, stream>>>(batch, offs);
  k_scatter<<<Nn, 128, 0, stream>>>(x, pe_re, pe_im, batch, offs, hd, arD, aiD, maskD, intra);
  k_pn<<<(Bb * Mm * Dd + 255) / 256, 256, 0, stream>>>(arD, aiD, pnD, Bb * Mm * Dd);

  // weight transposes -> bf16 [N][K]
  auto wt = [&](const float* s, __bf16* d, int K, int N) {
    k_wT<<<(K * N + 255) / 256, 256, 0, stream>>>(s, d, K, N);
  };
  wt(wq, wqT, Cc, Cc); wt(wk, wkT, Cc, Cc); wt(wv, wvT, Cc, Cc); wt(wo, woT, Cc, Cc);
  wt(pe_w1, w1T, 64, 32);
  wt(mlp_w1, m1T, Cc, 256);
  wt(mlp_w2, m2T, 256, Cc);
  k_w2pad<<<2, 256, 0, stream>>>(pe_w2, w2T);

  // QKV projections
  int BM = Bb * Mm;
  auto gemm = [&](const __bf16* A, const __bf16* BT, const float* bias, float* C,
                  int Mr, int N, int K, int relu) {
    k_gemm<<<(Mr / 16) * (N / 16), 32, 0, stream>>>(A, BT, bias, C, Mr, N, K, relu);
  };
  gemm(hd, wqT, bq, qF, BM, Cc, Cc, 0);
  gemm(hd, wkT, bk, kF, BM, Cc, Cc, 0);
  gemm(hd, wvT, bv, vF, BM, Cc, Cc, 0);
  k_vT<<<(Bb * Mm * Cc + 255) / 256, 256, 0, stream>>>(vF, vT);

  // pairwise PE-bias MLP (dominant compute): B*M*16 waves, 5 WMMAs each
  k_pair<<<Bb * Mm * 16, 32, 0, stream>>>(arD, aiD, pnD, w1T, w2T, pe_b1, pe_b2, biasA);

  // fused scores + softmax + AV
  k_attn<<<Bb * Hh * 16, 32, 0, stream>>>(qF, kF, maskD, biasA, vT, oF);

  // output projection, gather + residual + LN2
  k_f32_to_bf16<<<(BM * Cc + 255) / 256, 256, 0, stream>>>(oF, oB, BM * Cc);
  gemm(oB, woT, bo, oPrj, BM, Cc, Cc, 0);
  k_resln<<<Nn, 128, 0, stream>>>(oPrj, x, batch, intra, ln2_g, ln2_b, hF, hB);

  // FFN + residual + LN3
  gemm(hB, m1T, mlp_b1, f1F, Nn, 256, Cc, 1);
  k_f32_to_bf16<<<(Nn * 256 + 255) / 256, 256, 0, stream>>>(f1F, f1B, Nn * 256);
  gemm(f1B, m2T, mlp_b2, f2F, Nn, Cc, 256, 0);
  k_final<<<Nn, 128, 0, stream>>>(hF, f2F, ln3_g, ln3_b, outp);
}
